// LongSelfAttention_36009005809754
// MI455X (gfx1250) — matrix operand order repaired and checked by reading, not compiled, not run on hardware
//
#include <hip/hip_runtime.h>
#include <hip/hip_bf16.h>

// ---------------------------------------------------------------------------
// Longformer sliding-chunk attention for gfx1250 (MI455X), wave32 + WMMA f16.
// B=2, T=4096, E=768, H=12, s=64, w=256.
// - QKV projection: WMMA f32_16x16x32_f16 GEMM (A/B fragments are contiguous
//   per-lane loads thanks to C = X * W^T both being k-major).
// - Attention: fused flash-style, K/V tiles staged to LDS with CDNA5
//   async global->LDS DMA (ASYNCcnt), double buffered; P transposed to
//   A-layout through a per-wave LDS bounce.
// ---------------------------------------------------------------------------

typedef _Float16 f16;
typedef __attribute__((ext_vector_type(16))) _Float16 v16h;
typedef __attribute__((ext_vector_type(8)))  _Float16 v8h;
typedef __attribute__((ext_vector_type(8)))  float    v8f;
typedef __attribute__((ext_vector_type(4)))  int      v4i;

#define B_   2
#define T_   4096
#define E_   768
#define H_   12
#define S_   64
#define W_   256
#define TP_  (T_ + 2 * W_)          // 4608 padded key rows per (b,h)
#define NCH  (T_ / W_)              // 16 chunks
#define NEGV (-1.0e30f)

// f16 workspace layout (byte offsets); total 57,016,320 bytes.
#define XH_OFF  (size_t)0                       // X f16:  8192*768
#define WQ_OFF  (size_t)12582912                // 768*768 each
#define WK_OFF  (size_t)13762560
#define WV_OFF  (size_t)14942208
#define QH_OFF  (size_t)16121856                // (b,h,t,d)       2*12*4096*64
#define KH_OFF  (size_t)28704768                // (b,h,tpad,d)    2*12*4608*64
#define VT_OFF  (size_t)42860544                // (b,h,d,tpad)    2*12*64*4608
#define KH_BYTES (size_t)(2 * 12 * TP_ * 64 * 2)
#define VT_BYTES (size_t)(2 * 12 * 64 * TP_ * 2)

#if defined(__has_builtin)
#if __has_builtin(__builtin_amdgcn_global_load_async_to_lds_b128)
#define HAVE_ASYNC_LDS 1
#endif
#endif
#ifndef HAVE_ASYNC_LDS
#define HAVE_ASYNC_LDS 0
#endif

static __device__ __forceinline__ void async_cp16(const f16* g, f16* l) {
#if HAVE_ASYNC_LDS
  // param1: addrspace(1) int4*  (global src), param2: addrspace(3) int4* (LDS
  // dst), then immediate offset and cpol.
  __builtin_amdgcn_global_load_async_to_lds_b128(
      (__attribute__((address_space(1))) v4i*)g,
      (__attribute__((address_space(3))) v4i*)l, 0, 0);
#else
  *(v8h*)l = *(const v8h*)g;      // sync fallback: global_load + ds_store
#endif
}

static __device__ __forceinline__ void wait_async_lds() {
#if HAVE_ASYNC_LDS
#if __has_builtin(__builtin_amdgcn_s_wait_asynccnt)
  __builtin_amdgcn_s_wait_asynccnt(0);
#else
  asm volatile("s_wait_asynccnt 0" ::: "memory");
#endif
#endif
}

static __device__ __forceinline__ v16h make_afrag(const f16* __restrict__ p) {
  // A-matrix 16x32 f16 layout: per lane, two contiguous 8-half chunks at
  // k = hi*8 and k = 16 + hi*8 (caller bakes hi*8 into p).
  v8h a0 = *(const v8h*)(p);
  v8h a1 = *(const v8h*)(p + 16);
  v16h r;
#pragma unroll
  for (int i = 0; i < 8; ++i) { r[i] = a0[i]; r[i + 8] = a1[i]; }
  return r;
}

static __device__ __forceinline__ float red16_max(float v) {
  v = fmaxf(v, __shfl_xor(v, 1, 32));
  v = fmaxf(v, __shfl_xor(v, 2, 32));
  v = fmaxf(v, __shfl_xor(v, 4, 32));
  v = fmaxf(v, __shfl_xor(v, 8, 32));
  return v;
}
static __device__ __forceinline__ float red16_sum(float v) {
  v += __shfl_xor(v, 1, 32);
  v += __shfl_xor(v, 2, 32);
  v += __shfl_xor(v, 4, 32);
  v += __shfl_xor(v, 8, 32);
  return v;
}

// ---------------------------------------------------------------------------
__global__ void cast_f32_f16(const float* __restrict__ src,
                             f16* __restrict__ dst, int n) {
  int i = blockIdx.x * blockDim.x + threadIdx.x;
  if (i < n) dst[i] = (f16)src[i];
}

// ---------------------------------------------------------------------------
// QKV GEMM: C[m,n] = sum_k X[m,k] * W[n,k]   (M=8192, N=768, K=768)
// z: 0->Q (scaled 1/8, layout (b,h,t,d)), 1->K ((b,h,t+w,d)), 2->V^T ((b,h,d,t+w))
__global__ __launch_bounds__(256) void qkv_gemm(
    const f16* __restrict__ X,
    const f16* __restrict__ WQm, const f16* __restrict__ WKm,
    const f16* __restrict__ WVm,
    f16* __restrict__ Qh, f16* __restrict__ Kh, f16* __restrict__ Vt) {
  const int z    = blockIdx.z;
  const f16* Wm  = (z == 0) ? WQm : ((z == 1) ? WKm : WVm);
  const int lane = threadIdx.x & 31;
  const int wv   = threadIdx.x >> 5;
  const int lo   = lane & 15;
  const int hi   = lane >> 4;
  const int m0   = blockIdx.x * 128 + wv * 16;   // wave's 16 rows
  const int n0   = blockIdx.y * 64;              // block's 64 cols

  v8f acc[4] = {};
#pragma unroll 2
  for (int k0 = 0; k0 < E_; k0 += 32) {
    const f16* xp = X + (size_t)(m0 + lo) * E_ + k0 + hi * 8;
    v16h a = make_afrag(xp);
#pragma unroll
    for (int nt = 0; nt < 4; ++nt) {
      const int n = n0 + nt * 16 + lo;
      v16h b = make_afrag(Wm + (size_t)n * E_ + k0 + hi * 8);
      acc[nt] = __builtin_amdgcn_wmma_f32_16x16x32_f16(
          false, a, false, b, (short)0, acc[nt], false, false);
    }
  }

  const float scale = (z == 0) ? 0.125f : 1.0f;   // 1/sqrt(64)
#pragma unroll
  for (int nt = 0; nt < 4; ++nt) {
    const int n = n0 + nt * 16 + lo;
    const int hh = n >> 6, d = n & 63;
#pragma unroll
    for (int r = 0; r < 8; ++r) {
      const int m = m0 + r + 8 * hi;
      const int bb = m >> 12, t = m & (T_ - 1);
      const f16 v = (f16)(acc[nt][r] * scale);
      if (z == 0)
        Qh[(((size_t)bb * H_ + hh) * T_ + t) * S_ + d] = v;
      else if (z == 1)
        Kh[(((size_t)bb * H_ + hh) * TP_ + t + W_) * S_ + d] = v;
      else
        Vt[(((size_t)bb * H_ + hh) * S_ + d) * TP_ + t + W_] = v;
    }
  }
}

// ---------------------------------------------------------------------------
// Fused sliding-window attention. Grid (chunk=16, head=12, batch=2), 8 waves.
// Each wave carries TWO 16-row query tiles through one pass over 24 key-tiles
// of 32 keys; K/V tiles staged to LDS by async DMA, double buffered.
__global__ __launch_bounds__(256) void sliding_attn(
    const f16* __restrict__ Qh, const f16* __restrict__ Kh,
    const f16* __restrict__ Vt, float* __restrict__ out) {
  __shared__ __align__(16) f16 ldsK[2][32 * 64];   // (key,d) row-major, 2x4KB
  __shared__ __align__(16) f16 ldsV[2][64 * 32];   // (d,key) row-major, 2x4KB
  __shared__ __align__(16) f16 pbuf[8][2][16 * 32]; // per-wave P bounce, 16KB

  const int tid  = threadIdx.x;
  const int lane = tid & 31;
  const int wv   = tid >> 5;
  const int lo   = lane & 15;
  const int hi   = lane >> 4;
  const int c    = blockIdx.x;
  const int h    = blockIdx.y;
  const int bb   = blockIdx.z;

  const size_t qbase = (((size_t)bb * H_ + h) * T_ + (size_t)c * W_) * S_;
  // padded key row for attention column j is c*W_ + j  (key_abs=(c-1)w+j)
  const f16* Kg = Kh + ((size_t)bb * H_ + h) * TP_ * S_ + (size_t)c * W_ * S_;
  const f16* Vg = Vt + ((size_t)bb * H_ + h) * S_ * TP_ + (size_t)c * W_;

  // per-thread staging slices: K tile (32x64 f16) is 4KB contiguous;
  // V tile is 64 segments (d rows) of 64B.
  const int vseg = tid >> 2, vpart = tid & 3;

  // ---- load Q fragments for both row tiles --------------------------------
  const int mb0 = wv * 32, mb1 = wv * 32 + 16;
  v16h aq[2][2];
#pragma unroll
  for (int t2 = 0; t2 < 2; ++t2) {
    const f16* qp = Qh + qbase + (size_t)(wv * 32 + t2 * 16 + lo) * S_ + hi * 8;
    aq[t2][0] = make_afrag(qp);
    aq[t2][1] = make_afrag(qp + 32);
  }

  float rm[2][8], rs[2][8];
  v8f o[2][4] = {};
#pragma unroll
  for (int t2 = 0; t2 < 2; ++t2)
#pragma unroll
    for (int r = 0; r < 8; ++r) { rm[t2][r] = NEGV; rs[t2][r] = 0.0f; }

  // ---- prologue: stage tile kt=0 into buffer 0 ----------------------------
  async_cp16(Kg + tid * 8, &ldsK[0][tid * 8]);
  async_cp16(Vg + (size_t)vseg * TP_ + vpart * 8, &ldsV[0][vseg * 32 + vpart * 8]);
  wait_async_lds();
  __syncthreads();

  for (int kt = 0; kt < 24; ++kt) {
    const int cur = kt & 1, nxt = cur ^ 1;
    const int j0 = kt * 32;

    // issue async stage of next tile (overlaps with compute below)
    if (kt + 1 < 24) {
      const int jn = (kt + 1) * 32;
      async_cp16(Kg + (size_t)jn * S_ + tid * 8, &ldsK[nxt][tid * 8]);
      async_cp16(Vg + (size_t)vseg * TP_ + jn + vpart * 8,
                 &ldsV[nxt][vseg * 32 + vpart * 8]);
    }

    // ---- S = Q K^T, mask, online softmax, P -> pbuf (both row tiles) ------
#pragma unroll
    for (int t2 = 0; t2 < 2; ++t2) {
      const int m_base = (t2 == 0) ? mb0 : mb1;
      v8f s[2];
#pragma unroll
      for (int half = 0; half < 2; ++half) {
        const f16* kp = &ldsK[cur][(half * 16 + lo) * 64 + hi * 8];
        const v16h bk0 = make_afrag(kp);
        const v16h bk1 = make_afrag(kp + 32);
        v8f sv = {};
        sv = __builtin_amdgcn_wmma_f32_16x16x32_f16(false, aq[t2][0], false,
                                                    bk0, (short)0, sv, false, false);
        sv = __builtin_amdgcn_wmma_f32_16x16x32_f16(false, aq[t2][1], false,
                                                    bk1, (short)0, sv, false, false);
        s[half] = sv;
      }

      if (c == 0 || c == NCH - 1) {
#pragma unroll
        for (int half = 0; half < 2; ++half) {
          const int col = j0 + half * 16 + lo;
#pragma unroll
          for (int r = 0; r < 8; ++r) {
            const int rr = m_base + r + 8 * hi;
            if (c == 0       && col + rr < W_)          s[half][r] = NEGV;
            if (c == NCH - 1 && col + rr >= 3 * W_ - 1) s[half][r] = NEGV;
          }
        }
      }

      f16* pb = &pbuf[wv][t2][0];
#pragma unroll
      for (int r = 0; r < 8; ++r) {
        float tm = red16_max(fmaxf(s[0][r], s[1][r]));
        const float nm = fmaxf(rm[t2][r], tm);
        const float fs = __expf(rm[t2][r] - nm);
        const float p0 = __expf(s[0][r] - nm);
        const float p1 = __expf(s[1][r] - nm);
        rs[t2][r] = rs[t2][r] * fs + red16_sum(p0 + p1);
        rm[t2][r] = nm;
#pragma unroll
        for (int dt = 0; dt < 4; ++dt) o[t2][dt][r] *= fs;
        pb[(r + 8 * hi) * 32 + lo]      = (f16)p0;
        pb[(r + 8 * hi) * 32 + 16 + lo] = (f16)p1;
      }
    }
    __syncthreads();   // pbuf visible; all ldsK[cur] reads complete

    // ---- O += P V  (V B-fragments from LDS, shared by both row tiles) -----
    v16h ap[2];
#pragma unroll
    for (int t2 = 0; t2 < 2; ++t2) {
      const f16* pp = &pbuf[wv][t2][lo * 32 + hi * 8];
      v8h p0v = *(const v8h*)(pp);
      v8h p1v = *(const v8h*)(pp + 16);
#pragma unroll
      for (int i = 0; i < 8; ++i) { ap[t2][i] = p0v[i]; ap[t2][i + 8] = p1v[i]; }
    }
#pragma unroll
    for (int dt = 0; dt < 4; ++dt) {
      const v16h bv = make_afrag(&ldsV[cur][(dt * 16 + lo) * 32 + hi * 8]);
#pragma unroll
      for (int t2 = 0; t2 < 2; ++t2)
        o[t2][dt] = __builtin_amdgcn_wmma_f32_16x16x32_f16(
            false, ap[t2], false, bv, (short)0, o[t2][dt], false, false);
    }

    wait_async_lds();   // per-wave: own DMA done; barrier makes all visible
    __syncthreads();    // next buffers staged; pbuf reusable
  }

  // ---- normalize + store f32 output (b, t, h*64+d) ------------------------
#pragma unroll
  for (int t2 = 0; t2 < 2; ++t2) {
    const int m_base = (t2 == 0) ? mb0 : mb1;
#pragma unroll
    for (int r = 0; r < 8; ++r) {
      const float inv = 1.0f / rs[t2][r];
      const size_t t = (size_t)c * W_ + m_base + r + 8 * hi;
#pragma unroll
      for (int dt = 0; dt < 4; ++dt) {
        out[((size_t)bb * T_ + t) * E_ + h * S_ + dt * 16 + lo] =
            o[t2][dt][r] * inv;
      }
    }
  }
}

// ---------------------------------------------------------------------------
extern "C" void kernel_launch(void* const* d_in, const int* in_sizes, int n_in,
                              void* d_out, int out_size, void* d_ws, size_t ws_size,
                              hipStream_t stream) {
  const float* x  = (const float*)d_in[0];
  const float* wq = (const float*)d_in[1];
  const float* wk = (const float*)d_in[2];
  const float* wv = (const float*)d_in[3];

  char* ws = (char*)d_ws;                 // needs ~57 MB of scratch
  f16* xh  = (f16*)(ws + XH_OFF);
  f16* wqh = (f16*)(ws + WQ_OFF);
  f16* wkh = (f16*)(ws + WK_OFF);
  f16* wvh = (f16*)(ws + WV_OFF);
  f16* qh  = (f16*)(ws + QH_OFF);
  f16* kh  = (f16*)(ws + KH_OFF);
  f16* vt  = (f16*)(ws + VT_OFF);

  const int nx = B_ * T_ * E_;
  const int nw = E_ * E_;
  cast_f32_f16<<<(nx + 255) / 256, 256, 0, stream>>>(x, xh, nx);
  cast_f32_f16<<<(nw + 255) / 256, 256, 0, stream>>>(wq, wqh, nw);
  cast_f32_f16<<<(nw + 255) / 256, 256, 0, stream>>>(wk, wkh, nw);
  cast_f32_f16<<<(nw + 255) / 256, 256, 0, stream>>>(wv, wvh, nw);

  // zero K/V pads (w rows each side per (b,h)); cheap, graph-capturable
  (void)hipMemsetAsync(kh, 0, KH_BYTES, stream);
  (void)hipMemsetAsync(vt, 0, VT_BYTES, stream);

  qkv_gemm<<<dim3((B_ * T_) / 128, E_ / 64, 3), 256, 0, stream>>>(
      xh, wqh, wkh, wvh, qh, kh, vt);

  sliding_attn<<<dim3(NCH, H_, B_), 256, 0, stream>>>(
      qh, kh, vt, (float*)d_out);
}
